// LSTM_15522011807883
// MI455X (gfx1250) — matrix operand order repaired and checked by reading, not compiled, not run on hardware
//
#include <hip/hip_runtime.h>
#include <hip/hip_bf16.h>

typedef __attribute__((ext_vector_type(16))) __bf16 v16bf;
typedef __attribute__((ext_vector_type(8)))  float  v8f;

#define B_   64
#define T_   1024
#define I_   128
#define H_   512
#define HROW 520   // 512 + 8 pad to stagger LDS banks

__device__ __forceinline__ unsigned short f2bf(float f) {
    // round-to-nearest-even f32 -> bf16
    unsigned u = __builtin_bit_cast(unsigned, f);
    unsigned r = 0x7FFFu + ((u >> 16) & 1u);
    u += r;
    return (unsigned short)(u >> 16);
}
__device__ __forceinline__ float sigm(float v) {
    return 1.0f / (1.0f + __expf(-v));
}
__device__ __forceinline__ float tanh_fast(float v) {
    float e = __expf(2.0f * v);
    return 1.0f - 2.0f / (e + 1.0f);
}
// Defeat LICM hoist->spill of loop-invariant fragment loads: redefine the base
// pointer each iteration (stays in an SGPR pair; loads remain saddr-based).
__device__ __forceinline__ const uint4* launder(const uint4* p) {
    unsigned long long u = (unsigned long long)p;
    asm volatile("" : "+s"(u));
    return (const uint4*)u;
}

// ---- Pre-pack V [4,512,512] f32 -> bf16 in WMMA B-fragment blob order ----
// blob(ct,ks) = 1KB; lane L's 32B = bf16 V[g][ks*32 + (q>>3)*16 + (L>>4)*8 + (q&7)][ct*16 + (L&15)]
__global__ void pack_V_kernel(const float* __restrict__ V, unsigned short* __restrict__ Vp) {
    int e  = blockIdx.x * 256 + threadIdx.x;      // 0 .. 4*512*512-1
    int q  = e & 15;
    int ln = (e >> 4) & 31;
    int ks = (e >> 9) & 15;
    int ct = e >> 13;
    int g  = ct >> 5;
    int k  = ks * 32 + ((q >> 3) << 4) + ((ln >> 4) << 3) + (q & 7);
    int n  = ((ct & 31) << 4) + (ln & 15);
    Vp[e] = f2bf(V[((size_t)(g * 512 + k)) * 512 + n]);
}

// ---- Pre-pack U [4,128,512] likewise (4 K-blocks) ----
__global__ void pack_U_kernel(const float* __restrict__ U, unsigned short* __restrict__ Up) {
    int e  = blockIdx.x * 256 + threadIdx.x;      // 0 .. 4*128*512-1
    int q  = e & 15;
    int ln = (e >> 4) & 31;
    int ks = (e >> 9) & 3;
    int ct = e >> 11;
    int g  = ct >> 5;
    int k  = ks * 32 + ((q >> 3) << 4) + ((ln >> 4) << 3) + (q & 7);
    int n  = ((ct & 31) << 4) + (ln & 15);
    Up[e] = f2bf(U[((size_t)(g * 128 + k)) * 512 + n]);
}

// ---- Pre-pack x [64,1024,128] f32 -> bf16 in WMMA A-fragment blob order ----
// blob(bt,t,kb) = 1KB; lane L reg j pair s:
//   k = kb*32 + (j>>2)*16 + (L>>4)*8 + (j&3)*2 + s ;  b = bt*16 + (L&15)
__global__ void pack_X_kernel(const float* __restrict__ x, unsigned short* __restrict__ Xp) {
    int e    = blockIdx.x * 256 + threadIdx.x;    // 0 .. 64*1024*128-1
    int s    = e & 1;
    int j    = (e >> 1) & 7;
    int ln   = (e >> 4) & 31;
    int kb   = (e >> 9) & 3;
    int t    = (e >> 11) & 1023;
    int bt   = e >> 21;
    int m    = ln & 15;
    int half = ln >> 4;
    int k    = kb * 32 + ((j >> 2) << 4) + half * 8 + ((j & 3) << 1) + s;
    Xp[e] = f2bf(x[((size_t)(bt * 16 + m) * T_ + t) * I_ + k]);
}

// ---- Persistent LSTM kernel: 4 workgroups x 512 threads (16 waves) ----
// WG handles 16 batch rows. Wave w owns hidden cols [32w,32w+32) for all 4 gates
// => 8 output tiles of 16x16. c-state lives in registers; h exchanged via LDS.
__global__ void __launch_bounds__(512)
lstm_persistent(const unsigned short* __restrict__ Xp,
                const unsigned short* __restrict__ Up,
                const unsigned short* __restrict__ Vp,
                const float* __restrict__ bias,
                float* __restrict__ out)
{
    __shared__ __align__(16) unsigned short hbuf[2][16 * HROW];

    const int tid    = threadIdx.x;
    const int lane   = tid & 31;
    const int w      = tid >> 5;      // wave 0..15
    const int half   = lane >> 4;     // 0/1
    const int mlo    = lane & 15;
    const int wgBase = blockIdx.x * 16;

    // tile tt = g*2+jj -> column-tile id ct = g*32 + w*2 + jj  (n_full = ct*16)
    int   ct[8];
    float bval[8];
#pragma unroll
    for (int g = 0; g < 4; ++g)
#pragma unroll
        for (int jj = 0; jj < 2; ++jj) {
            int tt   = g * 2 + jj;
            ct[tt]   = g * 32 + w * 2 + jj;
            bval[tt] = bias[ct[tt] * 16 + mlo];
        }

    // zero the first h buffer (h0 = 0)
    for (int i = tid; i < 16 * HROW; i += 512) hbuf[0][i] = 0;
    __syncthreads();

    const uint4* Vp4 = (const uint4*)Vp;
    const uint4* Up4 = (const uint4*)Up;
    const uint4* Xp4 = (const uint4*)Xp;

    float* outH  = out;                                 // [B,T,H]
    float* outHT = out + (size_t)B_ * T_ * H_;          // [B,H]
    float* outCT = outHT + (size_t)B_ * H_;             // [B,H]

    float cst[2][8];
#pragma unroll
    for (int jj = 0; jj < 2; ++jj)
#pragma unroll
        for (int r = 0; r < 8; ++r) cst[jj][r] = 0.0f;

    int p = 0;

    for (int t = 0; t < T_; ++t) {
        const uint4* Up4t = launder(Up4);
        const uint4* Vp4t = launder(Vp4);

        // accumulators = bias broadcast
        v8f acc[8];
#pragma unroll
        for (int tt = 0; tt < 8; ++tt) {
            v8f a;
#pragma unroll
            for (int r = 0; r < 8; ++r) a[r] = bval[tt];
            acc[tt] = a;
        }

        // ---- x_t @ U : K = 128 = 4 blocks of 32, A pre-packed in Xp ----
        size_t xbase = (((size_t)blockIdx.x * T_ + t) * 4) * 64 + (size_t)lane * 2;
#pragma unroll
        for (int kb = 0; kb < 4; ++kb) {
            union { uint4 q[2]; v16bf v; } A;
            A.q[0] = Xp4[xbase + kb * 64];
            A.q[1] = Xp4[xbase + kb * 64 + 1];
#pragma unroll
            for (int tt = 0; tt < 8; ++tt) {
                int bi = (ct[tt] * 4 + kb) * 64 + lane * 2;
                union { uint4 q[2]; v16bf v; } Bf;
                Bf.q[0] = Up4t[bi];
                Bf.q[1] = Up4t[bi + 1];
                acc[tt] = __builtin_amdgcn_wmma_f32_16x16x32_bf16(
                    false, A.v, false, Bf.v, (short)0, acc[tt], false, false);
            }
        }

        // ---- h_{t-1} @ V : K = 512 = 16 blocks of 32 ----
        const uint4* hrow = (const uint4*)&hbuf[p][mlo * HROW];
        for (int ks = 0; ks < 16; ++ks) {
            union { uint4 q[2]; v16bf v; } A;
            int e0 = ks * 32 + half * 8;      // ushort offset in row
            A.q[0] = hrow[e0 >> 3];
            A.q[1] = hrow[(e0 + 16) >> 3];
#pragma unroll
            for (int tt = 0; tt < 8; ++tt) {
                int bi = (ct[tt] * 16 + ks) * 64 + lane * 2;
                union { uint4 q[2]; v16bf v; } Bf;
                Bf.q[0] = Vp4t[bi];
                Bf.q[1] = Vp4t[bi + 1];
                acc[tt] = __builtin_amdgcn_wmma_f32_16x16x32_bf16(
                    false, A.v, false, Bf.v, (short)0, acc[tt], false, false);
            }
        }

        // ---- gates, cell update, emit h ----
#pragma unroll
        for (int jj = 0; jj < 2; ++jj) {
#pragma unroll
            for (int r = 0; r < 8; ++r) {
                float iv = sigm(acc[0 + jj][r]);
                float fv = sigm(acc[2 + jj][r]);
                float gv = tanh_fast(acc[4 + jj][r]);
                float ov = sigm(acc[6 + jj][r]);
                float cc = fv * cst[jj][r] + iv * gv;
                cst[jj][r] = cc;
                float hv = ov * tanh_fast(cc);
                int m = half * 8 + r;
                int k = w * 32 + jj * 16 + mlo;
                hbuf[1 - p][m * HROW + k] = f2bf(hv);
                outH[((size_t)(wgBase + m) * T_ + t) * H_ + k] = hv;
                if (t == T_ - 1) {
                    outHT[(size_t)(wgBase + m) * H_ + k] = hv;
                    outCT[(size_t)(wgBase + m) * H_ + k] = cc;
                }
            }
        }
        __syncthreads();
        p ^= 1;
    }
}

extern "C" void kernel_launch(void* const* d_in, const int* in_sizes, int n_in,
                              void* d_out, int out_size, void* d_ws, size_t ws_size,
                              hipStream_t stream) {
    (void)in_sizes; (void)n_in; (void)out_size; (void)ws_size;
    const float* x = (const float*)d_in[0];
    const float* U = (const float*)d_in[1];
    const float* V = (const float*)d_in[2];
    const float* b = (const float*)d_in[3];

    unsigned short* Vp = (unsigned short*)d_ws;            // 4*512*512 bf16 = 2 MB
    unsigned short* Up = Vp + 4 * 512 * 512;               // 4*128*512 bf16 = 512 KB
    unsigned short* Xp = Up + 4 * 128 * 512;               // 64*1024*128 bf16 = 16 MB
    float* out = (float*)d_out;

    pack_V_kernel<<<(4 * 512 * 512) / 256, 256, 0, stream>>>(V, Vp);
    pack_U_kernel<<<(4 * 128 * 512) / 256, 256, 0, stream>>>(U, Up);
    pack_X_kernel<<<(64 * 1024 * 128) / 256, 256, 0, stream>>>(x, Xp);
    lstm_persistent<<<4, 512, 0, stream>>>(Xp, Up, Vp, b, out);
}
